// SeperatedBlockUpdateLayer_72937134620976
// MI455X (gfx1250) — matrix operand order repaired and checked by reading, not compiled, not run on hardware
//
#include <hip/hip_runtime.h>
#include <math.h>

typedef __bf16 bf16_t;
typedef __attribute__((ext_vector_type(16))) __bf16 v16bf;
typedef __attribute__((ext_vector_type(8)))  __bf16 v8bf;
typedef __attribute__((ext_vector_type(8)))  float  v8f;
typedef __attribute__((ext_vector_type(4)))  unsigned int u32x4;
typedef __attribute__((ext_vector_type(4)))  int i32x4;
typedef __attribute__((ext_vector_type(8)))  int i32x8;

#define CCH 64
#define NGRP 16384            // 2^20 rows / 64
#define BLK_ELEMS 262144      // 64ch * 64 * 64 per block

#ifndef __has_builtin
#define __has_builtin(x) 0
#endif

#if __has_builtin(__builtin_amdgcn_tensor_load_to_lds)
#define USE_TDM 1
#else
#define USE_TDM 0
#endif

__device__ inline void wait_tensorcnt0() {
#if __has_builtin(__builtin_amdgcn_s_wait_tensorcnt)
  __builtin_amdgcn_s_wait_tensorcnt(0);
#else
  asm volatile("s_wait_tensorcnt 0" ::: "memory");
#endif
}

#if USE_TDM
// TDM: DMA a 64x64 2-byte 2D tile (row stride 64 elems) from global into LDS,
// padding each 128B row with 16B so LDS rows land on a 72-element (144B) pitch.
// D# per CDNA5 ISA ch.8: group0 = {count, lds_addr, global_addr, type=2},
// group1 = {data_size=2B, pad_en, pad_interval=32dw, pad_amount=4dw, dims}.
__device__ inline void tdm_load_tile64(const void* gsrc, unsigned lds_off) {
  unsigned long long ga = (unsigned long long)gsrc;
  u32x4 g0;
  g0[0] = 1u;                                                // count=1 (user descriptor)
  g0[1] = lds_off;                                           // lds_addr (bytes)
  g0[2] = (unsigned)ga;                                      // global_addr[31:0]
  g0[3] = (unsigned)((ga >> 32) & 0x1FFFFFFu) | (2u << 30);  // addr[56:32] | type=2
  i32x8 g1;
  g1[0] = (1 << 16) | (1 << 20) | (4 << 22) | (3 << 25);     // data_size=2B, pad_enable, intv, amt
  g1[1] = (64 << 16);                                        // tensor_dim0 = 64 (bits 79:48)
  g1[2] = (64 << 16);                                        // tensor_dim1 = 64 (bits 111:80)
  g1[3] = (64 << 16);                                        // tile_dim0 = 64 (bits 127:112)
  g1[4] = 64;                                                // tile_dim1 = 64, tile_dim2 = 0
  g1[5] = 64;                                                // tensor_dim0_stride = 64
  g1[6] = 0;
  g1[7] = 0;
  i32x4 z4 = {0, 0, 0, 0};
#if defined(__clang_major__) && (__clang_major__ >= 23)
  i32x8 z8 = {0, 0, 0, 0, 0, 0, 0, 0};
  __builtin_amdgcn_tensor_load_to_lds(g0, g1, z4, z4, z8, 0);
#else
  __builtin_amdgcn_tensor_load_to_lds(g0, g1, z4, z4, 0);
#endif
}
#endif

__device__ inline v8f vzero8() {
  v8f z;
#pragma unroll
  for (int i = 0; i < 8; ++i) z[i] = 0.f;
  return z;
}

__device__ inline v8f wmma_bf16(v16bf a, v16bf b, v8f c) {
  return __builtin_amdgcn_wmma_f32_16x16x32_bf16(false, a, false, b, (short)0, c, false, false);
}

// A-operand (16x32 bf16): lane l holds row m=l&15; K chunk base kt + (l>=16?8:0), plus +16.
__device__ inline v16bf load_a_frag(const bf16_t* A, int lda, int lane, int mstrip, int kt) {
  int m = mstrip * 16 + (lane & 15);
  int kb = kt + ((lane >> 4) ? 8 : 0);
  const bf16_t* p = A + m * lda + kb;
  v8bf lo = *(const v8bf*)p;
  v8bf hi = *(const v8bf*)(p + 16);
  v16bf r;
#pragma unroll
  for (int i = 0; i < 8; ++i) { r[i] = lo[i]; r[8 + i] = hi[i]; }
  return r;
}

// B-operand (32x16 bf16): lane l holds col n=nb+(l&15); K chunk base kt + (l>=16?8:0), plus +16.
__device__ inline v16bf load_b_frag(const bf16_t* B, int ldb, int lane, int nb, int kt) {
  int n = nb + (lane & 15);
  int kc = kt + ((lane >> 4) ? 8 : 0);
  const bf16_t* p = B + n * ldb + kc;
  v8bf lo = *(const v8bf*)p;
  v8bf hi = *(const v8bf*)(p + 16);
  v16bf r;
#pragma unroll
  for (int i = 0; i < 8; ++i) { r[i] = lo[i]; r[8 + i] = hi[i]; }
  return r;
}

__device__ inline void gemm_rowmajor(const bf16_t* A, int lda, const bf16_t* B, int ldb,
                                     int Kdim, int lane, int mstrip, int nb0,
                                     v8f& acc0, v8f& acc1) {
  for (int kt = 0; kt < Kdim; kt += 32) {
    v16bf a  = load_a_frag(A, lda, lane, mstrip, kt);
    v16bf b0 = load_b_frag(B, ldb, lane, nb0,      kt);
    v16bf b1 = load_b_frag(B, ldb, lane, nb0 + 16, kt);
    acc0 = wmma_bf16(a, b0, acc0);
    acc1 = wmma_bf16(a, b1, acc1);
  }
}

// C/D layout: VGPR p of lane l -> (m = mstrip*16 + p + 8*(l>>4), n = nb + (l&15))
__device__ inline void stage_tile_lds(bf16_t* T, int ldt, int lane, int mstrip, int nb,
                                      const v8f& acc, const float* bias, bool do_relu) {
  int n  = nb + (lane & 15);
  float bv = bias ? bias[n] : 0.f;
  int m0 = mstrip * 16 + ((lane >> 4) ? 8 : 0);
#pragma unroll
  for (int p = 0; p < 8; ++p) {
    float v = acc[p] + bv;
    if (do_relu) v = fmaxf(v, 0.f);
    T[(m0 + p) * ldt + n] = (bf16_t)v;
  }
}

// Packed 16B store of 8 consecutive "m" (second-spatial) values into [b][c][s1][m] layout.
__device__ inline void store_m_global(bf16_t* dst, int s1, int lane, int mstrip, int nb,
                                      const v8f& acc, const float* bias) {
  int c  = nb + (lane & 15);
  int m0 = mstrip * 16 + ((lane >> 4) ? 8 : 0);
  float bv = bias[c];
  v8bf pk;
#pragma unroll
  for (int p = 0; p < 8; ++p) {
    float v = fmaxf(acc[p] + bv, 0.f);
    pk[p] = (bf16_t)v;
  }
  *(v8bf*)(dst + (((size_t)c << 6) + (size_t)s1) * 64 + m0) = pk;
}

__device__ inline void f32x16_to_lds(const float* __restrict__ src, bf16_t* dst) {
#pragma unroll
  for (int h = 0; h < 2; ++h) {
    v8bf pk;
#pragma unroll
    for (int q = 0; q < 8; ++q) pk[q] = (bf16_t)src[h * 8 + q];
    *(v8bf*)(dst + h * 8) = pk;
  }
}

__device__ inline void load_w_lds(const float* __restrict__ W, bf16_t* Ws, int tid) {
  int e = tid * 16;                 // 256 threads * 16 = 4096 elems
  int o = e >> 6, k = e & 63;
  f32x16_to_lds(W + e, Ws + o * 72 + k);
}

// ---------------- K1: four chained 64x64x64 GEMMs (the two 2-layer MLPs) ----------------
__global__ __launch_bounds__(256)
void k1_mlp(const float* __restrict__ x,
            const float* __restrict__ w11, const float* __restrict__ b11,
            const float* __restrict__ w12, const float* __restrict__ b12,
            const float* __restrict__ w21, const float* __restrict__ b21,
            const float* __restrict__ w22, const float* __restrict__ b22,
            bf16_t* __restrict__ m1t, bf16_t* __restrict__ m2t) {
  __shared__ bf16_t Xs[64 * 72];
  __shared__ bf16_t W11s[64 * 72], W12s[64 * 72], W21s[64 * 72], W22s[64 * 72];
  __shared__ bf16_t Ts[64 * 72];
  __shared__ float  Bs[4 * 64];
  int tid = threadIdx.x, lane = tid & 31, wave = tid >> 5;
  int g = blockIdx.x;               // row group: rows [g*64, g*64+64)
  int b = g >> 6, s1 = g & 63;
  {
    int e = tid * 16, i = e >> 6, k = e & 63;
    f32x16_to_lds(x + (size_t)g * 4096 + e, Xs + i * 72 + k);
  }
  load_w_lds(w11, W11s, tid);
  load_w_lds(w12, W12s, tid);
  load_w_lds(w21, W21s, tid);
  load_w_lds(w22, W22s, tid);
  if (tid < 64) {
    Bs[tid] = b11[tid]; Bs[64 + tid] = b12[tid];
    Bs[128 + tid] = b21[tid]; Bs[192 + tid] = b22[tid];
  }
  __syncthreads();
  int mstrip = wave & 3, nb0 = (wave >> 2) * 32;

  v8f a0 = vzero8(), a1 = vzero8();
  gemm_rowmajor(Xs, 72, W11s, 72, 64, lane, mstrip, nb0, a0, a1);   // t1 = relu(x W11^T + b11)
  stage_tile_lds(Ts, 72, lane, mstrip, nb0,      a0, Bs,       true);
  stage_tile_lds(Ts, 72, lane, mstrip, nb0 + 16, a1, Bs,       true);
  __syncthreads();

  a0 = vzero8(); a1 = vzero8();
  gemm_rowmajor(Ts, 72, W12s, 72, 64, lane, mstrip, nb0, a0, a1);   // m1 = relu(t1 W12^T + b12)
  bf16_t* d1 = m1t + (size_t)b * BLK_ELEMS;
  store_m_global(d1, s1, lane, mstrip, nb0,      a0, Bs + 64);
  store_m_global(d1, s1, lane, mstrip, nb0 + 16, a1, Bs + 64);
  __syncthreads();

  a0 = vzero8(); a1 = vzero8();
  gemm_rowmajor(Xs, 72, W21s, 72, 64, lane, mstrip, nb0, a0, a1);   // t2 = relu(x W21^T + b21)
  stage_tile_lds(Ts, 72, lane, mstrip, nb0,      a0, Bs + 128, true);
  stage_tile_lds(Ts, 72, lane, mstrip, nb0 + 16, a1, Bs + 128, true);
  __syncthreads();

  a0 = vzero8(); a1 = vzero8();
  gemm_rowmajor(Ts, 72, W22s, 72, 64, lane, mstrip, nb0, a0, a1);   // m2 = relu(t2 W22^T + b22)
  bf16_t* d2 = m2t + (size_t)b * BLK_ELEMS;
  store_m_global(d2, s1, lane, mstrip, nb0,      a0, Bs + 192);
  store_m_global(d2, s1, lane, mstrip, nb0 + 16, a1, Bs + 192);
}

// ---------------- K2: per-(block,channel) 64x64x64 correlation matmul ----------------
__global__ __launch_bounds__(128)
void k2_corr(const bf16_t* __restrict__ m1t, const bf16_t* __restrict__ m2t,
             bf16_t* __restrict__ hbuf) {
  __shared__ bf16_t As[64 * 72];
  __shared__ bf16_t Braw[64 * 72];
  __shared__ bf16_t Bt[64 * 72];
  int tid = threadIdx.x, lane = tid & 31, wave = tid >> 5;
  int g = blockIdx.x;               // g = b*64 + c
  int b = g >> 6, c = g & 63;
  const bf16_t* Ag = m1t + (size_t)g * 4096;   // [i][k] row-major
  const bf16_t* Bg = m2t + (size_t)g * 4096;   // [k][j] row-major
#if USE_TDM
  if (wave == 0) {
    tdm_load_tile64(Ag, (unsigned)(unsigned long long)(void*)As);
    tdm_load_tile64(Bg, (unsigned)(unsigned long long)(void*)Braw);
    wait_tensorcnt0();
  }
  __syncthreads();
#else
#pragma unroll
  for (int r = 0; r < 2; ++r) {                // A + Braw manual copy fallback
    int e = tid * 32 + r * 16;
    int i = e >> 6, k = e & 63;
    *(v8bf*)(As + i * 72 + k)       = *(const v8bf*)(Ag + e);
    *(v8bf*)(As + i * 72 + k + 8)   = *(const v8bf*)(Ag + e + 8);
    *(v8bf*)(Braw + i * 72 + k)     = *(const v8bf*)(Bg + e);
    *(v8bf*)(Braw + i * 72 + k + 8) = *(const v8bf*)(Bg + e + 8);
  }
  __syncthreads();
#endif
  {                                            // B: transpose Braw[k][j] -> Bt[j][k]
    int k = tid >> 1, jb = (tid & 1) * 32;
    const bf16_t* row = Braw + k * 72 + jb;
#pragma unroll
    for (int q = 0; q < 4; ++q) {
      v8bf v = *(const v8bf*)(row + q * 8);
#pragma unroll
      for (int e2 = 0; e2 < 8; ++e2) Bt[(jb + q * 8 + e2) * 72 + k] = v[e2];
    }
  }
  __syncthreads();
  int mstrip = wave;                           // 4 waves, one 16-row i-strip each
  v8f acc[4];
#pragma unroll
  for (int nt = 0; nt < 4; ++nt) acc[nt] = vzero8();
  for (int kt = 0; kt < 64; kt += 32) {
    v16bf a = load_a_frag(As, 72, lane, mstrip, kt);
#pragma unroll
    for (int nt = 0; nt < 4; ++nt) {
      v16bf bb = load_b_frag(Bt, 72, lane, nt * 16, kt);
      acc[nt] = wmma_bf16(a, bb, acc[nt]);
    }
  }
  bf16_t* hb = hbuf + (size_t)b * BLK_ELEMS + c;   // h[(b,i,j)][c]
  int m0 = mstrip * 16 + ((lane >> 4) ? 8 : 0);
#pragma unroll
  for (int nt = 0; nt < 4; ++nt) {
    int j = nt * 16 + (lane & 15);
#pragma unroll
    for (int p = 0; p < 8; ++p) {
      int i = m0 + p;
      float v = sqrtf(fmaxf(acc[nt][p], 0.f));
      hb[(size_t)(i * 64 + j) * 64] = (bf16_t)v;
    }
  }
}

// ---------------- K3: z1 = [x|h] @ uW1 + ub1, bf16 out + BN partial stats ----------------
__global__ __launch_bounds__(256)
void k3_lin1(const float* __restrict__ x, const bf16_t* __restrict__ hbuf,
             const float* __restrict__ uW1, const float* __restrict__ ub1,
             bf16_t* __restrict__ z1, float* __restrict__ partials) {
  __shared__ bf16_t As[64 * 136];
  __shared__ bf16_t Ws[64 * 136];
  __shared__ bf16_t Zs[64 * 72];
  __shared__ float  Bu[64];
  __shared__ float  Ps[4 * 64], Pq[4 * 64];
  int tid = threadIdx.x, lane = tid & 31, wave = tid >> 5;
  int g = blockIdx.x;
  size_t rowbase = (size_t)g * 64;
  {
    int e = tid * 16, i = e >> 6, k = e & 63;
    f32x16_to_lds(x + rowbase * 64 + e, As + i * 136 + k);          // cols 0..63: x
    const v8bf* src = (const v8bf*)(hbuf + rowbase * 64 + e);       // cols 64..127: h
    *(v8bf*)(As + i * 136 + 64 + k)     = src[0];
    *(v8bf*)(As + i * 136 + 64 + k + 8) = src[1];
  }
  {                                           // Ws[o][k] = uW1[k][o]  (uW1 is 128x64)
    int kk = tid >> 1, ob = (tid & 1) * 32;
    const float* row = uW1 + kk * 64 + ob;
#pragma unroll
    for (int q = 0; q < 32; ++q) Ws[(ob + q) * 136 + kk] = (bf16_t)row[q];
  }
  if (tid < 64) Bu[tid] = ub1[tid];
  __syncthreads();
  int mstrip = wave & 3, nb0 = (wave >> 2) * 32;
  v8f a0 = vzero8(), a1 = vzero8();
  gemm_rowmajor(As, 136, Ws, 136, 128, lane, mstrip, nb0, a0, a1);
  stage_tile_lds(Zs, 72, lane, mstrip, nb0,      a0, Bu, false);
  stage_tile_lds(Zs, 72, lane, mstrip, nb0 + 16, a1, Bu, false);
  __syncthreads();
  {                                           // coalesced z1 store
    int e = tid * 16, i = e >> 6, k = e & 63;
    *(v8bf*)(z1 + rowbase * 64 + e)     = *(const v8bf*)(Zs + i * 72 + k);
    *(v8bf*)(z1 + rowbase * 64 + e + 8) = *(const v8bf*)(Zs + i * 72 + k + 8);
  }
  {                                           // per-channel sum / sumsq
    int cc = tid & 63, q = tid >> 6;
    float S = 0.f, Q = 0.f;
#pragma unroll
    for (int r = 0; r < 16; ++r) {
      float v = (float)Zs[(q * 16 + r) * 72 + cc];
      S += v; Q += v * v;
    }
    Ps[q * 64 + cc] = S; Pq[q * 64 + cc] = Q;
  }
  __syncthreads();
  if (tid < 64) {
    float S = Ps[tid] + Ps[64 + tid] + Ps[128 + tid] + Ps[192 + tid];
    float Q = Pq[tid] + Pq[64 + tid] + Pq[128 + tid] + Pq[192 + tid];
    partials[(size_t)g * 128 + tid]      = S;
    partials[(size_t)g * 128 + 64 + tid] = Q;
  }
}

// ---------------- K4/K6: deterministic reduction -> per-channel BN scale/shift ----------------
__global__ __launch_bounds__(256)
void k_reduce_stats(const float* __restrict__ partials, const float* __restrict__ gamma,
                    const float* __restrict__ beta, float* __restrict__ sc,
                    int ngrp, float invN) {
  __shared__ float Rs[256], Rq[256];
  int c = blockIdx.x, t = threadIdx.x;
  float S = 0.f, Q = 0.f;
  for (int i = t; i < ngrp; i += 256) {
    S += partials[(size_t)i * 128 + c];
    Q += partials[(size_t)i * 128 + 64 + c];
  }
  Rs[t] = S; Rq[t] = Q;
  __syncthreads();
  for (int s = 128; s > 0; s >>= 1) {
    if (t < s) { Rs[t] += Rs[t + s]; Rq[t] += Rq[t + s]; }
    __syncthreads();
  }
  if (t == 0) {
    float mu  = Rs[0] * invN;
    float var = Rq[0] * invN - mu * mu;
    float rstd = rsqrtf(var + 1e-5f);
    float scale = gamma[c] * rstd;
    sc[c]      = scale;
    sc[64 + c] = beta[c] - mu * scale;
  }
}

// ---------------- K5: a1 = relu(bn1(z1)); z2 = a1 @ uW2 + ub2 + stats ----------------
__global__ __launch_bounds__(256)
void k5_lin2(const bf16_t* __restrict__ z1, const float* __restrict__ sc1,
             const float* __restrict__ uW2, const float* __restrict__ ub2,
             bf16_t* __restrict__ z2, float* __restrict__ partials) {
  __shared__ bf16_t As[64 * 72];
  __shared__ bf16_t Ws[64 * 72];
  __shared__ bf16_t Zs[64 * 72];
  __shared__ float  Sc[128];
  __shared__ float  Bu[64];
  __shared__ float  Ps[4 * 64], Pq[4 * 64];
  int tid = threadIdx.x, lane = tid & 31, wave = tid >> 5;
  int g = blockIdx.x;
  size_t rowbase = (size_t)g * 64;
  if (tid < 128) Sc[tid] = sc1[tid];
  if (tid < 64)  Bu[tid] = ub2[tid];
  __syncthreads();
  {
    int e = tid * 16, i = e >> 6, k = e & 63;
    const bf16_t* src = z1 + rowbase * 64 + e;
#pragma unroll
    for (int h = 0; h < 2; ++h) {
      v8bf in = *(const v8bf*)(src + h * 8);
      v8bf pk;
#pragma unroll
      for (int q = 0; q < 8; ++q) {
        int cc = k + h * 8 + q;
        float v = fmaxf((float)in[q] * Sc[cc] + Sc[64 + cc], 0.f);
        pk[q] = (bf16_t)v;
      }
      *(v8bf*)(As + i * 72 + k + h * 8) = pk;
    }
  }
  {                                           // Ws[o][k] = uW2[k][o] (uW2 is 64x64)
    int e = tid * 16, kk = e >> 6, ob = e & 63;
    const float* row = uW2 + e;
#pragma unroll
    for (int q = 0; q < 16; ++q) Ws[(ob + q) * 72 + kk] = (bf16_t)row[q];
  }
  __syncthreads();
  int mstrip = wave & 3, nb0 = (wave >> 2) * 32;
  v8f a0 = vzero8(), a1 = vzero8();
  gemm_rowmajor(As, 72, Ws, 72, 64, lane, mstrip, nb0, a0, a1);
  stage_tile_lds(Zs, 72, lane, mstrip, nb0,      a0, Bu, false);
  stage_tile_lds(Zs, 72, lane, mstrip, nb0 + 16, a1, Bu, false);
  __syncthreads();
  {
    int e = tid * 16, i = e >> 6, k = e & 63;
    *(v8bf*)(z2 + rowbase * 64 + e)     = *(const v8bf*)(Zs + i * 72 + k);
    *(v8bf*)(z2 + rowbase * 64 + e + 8) = *(const v8bf*)(Zs + i * 72 + k + 8);
  }
  {
    int cc = tid & 63, q = tid >> 6;
    float S = 0.f, Q = 0.f;
#pragma unroll
    for (int r = 0; r < 16; ++r) {
      float v = (float)Zs[(q * 16 + r) * 72 + cc];
      S += v; Q += v * v;
    }
    Ps[q * 64 + cc] = S; Pq[q * 64 + cc] = Q;
  }
  __syncthreads();
  if (tid < 64) {
    float S = Ps[tid] + Ps[64 + tid] + Ps[128 + tid] + Ps[192 + tid];
    float Q = Pq[tid] + Pq[64 + tid] + Pq[128 + tid] + Pq[192 + tid];
    partials[(size_t)g * 128 + tid]      = S;
    partials[(size_t)g * 128 + 64 + tid] = Q;
  }
}

// ---------------- K7: out = relu(bn2(z2)) + x ----------------
__global__ __launch_bounds__(256)
void k7_final(const float* __restrict__ x, const bf16_t* __restrict__ z2,
              const float* __restrict__ sc2, float* __restrict__ out) {
  __shared__ float Sc[128];
  int tid = threadIdx.x;
  if (tid < 128) Sc[tid] = sc2[tid];
  __syncthreads();
  size_t e = ((size_t)blockIdx.x * 256 + tid) * 16;
  const v8bf*   zp = (const v8bf*)(z2 + e);
  const float4* xp = (const float4*)(x + e);
  float4*       op = (float4*)(out + e);
  int cbase = (int)(e & 63);
#pragma unroll
  for (int h = 0; h < 2; ++h) {
    v8bf zv = zp[h];
#pragma unroll
    for (int q = 0; q < 2; ++q) {
      float4 xv = xp[h * 2 + q];
      float4 ov;
      int c0 = cbase + h * 8 + q * 4;
      ov.x = fmaxf((float)zv[q * 4 + 0] * Sc[c0 + 0] + Sc[64 + c0 + 0], 0.f) + xv.x;
      ov.y = fmaxf((float)zv[q * 4 + 1] * Sc[c0 + 1] + Sc[64 + c0 + 1], 0.f) + xv.y;
      ov.z = fmaxf((float)zv[q * 4 + 2] * Sc[c0 + 2] + Sc[64 + c0 + 2], 0.f) + xv.z;
      ov.w = fmaxf((float)zv[q * 4 + 3] * Sc[c0 + 3] + Sc[64 + c0 + 3], 0.f) + xv.w;
      op[h * 2 + q] = ov;
    }
  }
}

extern "C" void kernel_launch(void* const* d_in, const int* in_sizes, int n_in,
                              void* d_out, int out_size, void* d_ws, size_t ws_size,
                              hipStream_t stream) {
  const float* x   = (const float*)d_in[0];
  const float* w11 = (const float*)d_in[1];
  const float* b11 = (const float*)d_in[2];
  const float* w12 = (const float*)d_in[3];
  const float* b12 = (const float*)d_in[4];
  const float* w21 = (const float*)d_in[5];
  const float* b21 = (const float*)d_in[6];
  const float* w22 = (const float*)d_in[7];
  const float* b22 = (const float*)d_in[8];
  const float* uW1 = (const float*)d_in[9];
  const float* ub1 = (const float*)d_in[10];
  const float* g1  = (const float*)d_in[11];
  const float* be1 = (const float*)d_in[12];
  const float* uW2 = (const float*)d_in[13];
  const float* ub2 = (const float*)d_in[14];
  const float* g2  = (const float*)d_in[15];
  const float* be2 = (const float*)d_in[16];

  char* ws = (char*)d_ws;
  const size_t MB128 = 1ull << 27;
  bf16_t* m1t  = (bf16_t*)(ws);                 // [b][c][i][k], 128 MB
  bf16_t* m2t  = (bf16_t*)(ws + MB128);         // [b][c][k][j], 128 MB
  bf16_t* hbuf = (bf16_t*)(ws + 2 * MB128);     // [row][c],     128 MB
  bf16_t* z1   = (bf16_t*)(ws);                 // reuses m1t after K2
  bf16_t* z2   = (bf16_t*)(ws + MB128);         // reuses m2t after K2
  const size_t PSZ = (size_t)NGRP * 128 * sizeof(float);   // 8 MB each
  float* p1  = (float*)(ws + 3 * MB128);
  float* p2  = (float*)(ws + 3 * MB128 + PSZ);
  float* sc1 = (float*)(ws + 3 * MB128 + 2 * PSZ);
  float* sc2 = sc1 + 128;

  const float invN = 1.f / 1048576.f;

  k1_mlp<<<NGRP, 256, 0, stream>>>(x, w11, b11, w12, b12, w21, b21, w22, b22, m1t, m2t);
  k2_corr<<<NGRP, 128, 0, stream>>>(m1t, m2t, hbuf);
  k3_lin1<<<NGRP, 256, 0, stream>>>(x, hbuf, uW1, ub1, z1, p1);
  k_reduce_stats<<<64, 256, 0, stream>>>(p1, g1, be1, sc1, NGRP, invN);
  k5_lin2<<<NGRP, 256, 0, stream>>>(z1, sc1, uW2, ub2, z2, p2);
  k_reduce_stats<<<64, 256, 0, stream>>>(p2, g2, be2, sc2, NGRP, invN);
  k7_final<<<NGRP, 256, 0, stream>>>(x, z2, sc2, (float*)d_out);
}
